// NNConvProt_42021960024101
// MI455X (gfx1250) — matrix operand order repaired and checked by reading, not compiled, not run on hardware
//
#include <hip/hip_runtime.h>
#include <hip/hip_bf16.h>

// ---------------------------------------------------------------------------
// NNConv GNN forward, fused for MI455X (gfx1250, wave32, WMMA f16->f32).
//
// Core trick: per-edge message  msg_o = sum_{k,i} h_k * x_i * wb[k, i*OUT+o]
// is rewritten as  msg = (h (x) x_src) @ W2  with W2[k*IN+i, o] = wb[k, i*OUT+o],
// augmented with K rows for the theta-bias (x_src @ bb).  That is a plain GEMM
// tiled 16 edges x OUT with v_wmma_f32_16x16x32_f16 along K.
// ---------------------------------------------------------------------------

typedef __attribute__((ext_vector_type(16))) _Float16 v16h;
typedef __attribute__((ext_vector_type(8)))  _Float16 v8h;
typedef __attribute__((ext_vector_type(8)))  float    v8f;

#define NN 100000   // nodes
#define EE 200000   // edges
#define GG 2500     // graphs
// conv1: IN=16 OUT=32, Ktot = 16*16 + 16(bias) = 272 -> padded 288 (9 k-steps)
#define K1 288
// conv2: IN=32 OUT=16, Ktot = 16*32 + 32(bias) = 544 (17 k-steps exactly)
#define K2 544

static __device__ __forceinline__ v8f wmma16(v16h a, v16h b, v8f c) {
  // D = A(16x32 f16) * B(32x16 f16) + C(16x16 f32)
  return __builtin_amdgcn_wmma_f32_16x16x32_f16(false, a, false, b, (short)0, c,
                                                false, false);
}

// load one B-matrix fragment column from LDS (column-major, 16 contiguous
// halves starting at 32B-aligned offset) -> two 16-byte ds loads
static __device__ __forceinline__ v16h load_bcol(const _Float16* p) {
  v8h lo = *(const v8h*)(p);
  v8h hi = *(const v8h*)(p + 8);
  v16h r;
#pragma unroll
  for (int t = 0; t < 8; ++t) { r[t] = lo[t]; r[t + 8] = hi[t]; }
  return r;
}

// monotone float<->uint mapping for atomicMax on floats (incl. negatives)
static __device__ __forceinline__ unsigned enc_f(float f) {
  unsigned u = __float_as_uint(f);
  return (u & 0x80000000u) ? ~u : (u | 0x80000000u);
}
static __device__ __forceinline__ float dec_f(unsigned u) {
  return __uint_as_float((u & 0x80000000u) ? (u ^ 0x80000000u) : ~u);
}

// ---------------------------------------------------------------------------
// prep: build permuted f16 edge-weight matrices, column-major with bias rows.
//   W21[o*288 + r] : r<256 -> w_e1b[k*512 + i*32 + o] (r=k*16+i)
//                    r<272 -> b_e1b[(r-256)*32 + o], else 0
//   W22[o*544 + r] : r<512 -> w_e2b[k*512 + i*16 + o] (r=k*32+i)
//                    else  -> b_e2b[(r-512)*16 + o]
// ---------------------------------------------------------------------------
__global__ void prep_kernel(const float* __restrict__ w_e1b, const float* __restrict__ b_e1b,
                            const float* __restrict__ w_e2b, const float* __restrict__ b_e2b,
                            _Float16* __restrict__ W21, _Float16* __restrict__ W22) {
  int tid = blockIdx.x * blockDim.x + threadIdx.x;
  int stride = gridDim.x * blockDim.x;
  for (int idx = tid; idx < K1 * 32; idx += stride) {
    int o = idx / K1, r = idx % K1;
    float v;
    if (r < 256)      { int k = r >> 4, i = r & 15; v = w_e1b[k * 512 + i * 32 + o]; }
    else if (r < 272) { v = b_e1b[(r - 256) * 32 + o]; }
    else              { v = 0.0f; }
    W21[idx] = (_Float16)v;
  }
  for (int idx = tid; idx < K2 * 16; idx += stride) {
    int o = idx / K2, r = idx % K2;
    float v;
    if (r < 512) { int k = r >> 5, i = r & 31; v = w_e2b[k * 512 + i * 16 + o]; }
    else         { v = b_e2b[(r - 512) * 16 + o]; }
    W22[idx] = (_Float16)v;
  }
}

// ---------------------------------------------------------------------------
// gate: per-node attention gate + per-graph float atomicMax + node count
// ---------------------------------------------------------------------------
__global__ void gate_kernel(const float* __restrict__ x,
                            const float* __restrict__ w_g1, const float* __restrict__ b_g1,
                            const float* __restrict__ w_g2, const float* __restrict__ b_g2,
                            const int* __restrict__ batch,
                            float* __restrict__ gate, unsigned* __restrict__ gmax,
                            float* __restrict__ cnt) {
  int n = blockIdx.x * blockDim.x + threadIdx.x;
  if (n >= NN) return;
  float xr[16];
#pragma unroll
  for (int i = 0; i < 16; ++i) xr[i] = x[n * 16 + i];
  float g = b_g2[0];
#pragma unroll
  for (int k = 0; k < 16; ++k) {
    float acc = b_g1[k];
#pragma unroll
    for (int i = 0; i < 16; ++i) acc += xr[i] * w_g1[i * 16 + k];
    g += fmaxf(acc, 0.0f) * w_g2[k];
  }
  gate[n] = g;
  int gi = batch[n];
  atomicMax(&gmax[gi], enc_f(g));
  atomicAdd(&cnt[gi], 1.0f);
}

// ---------------------------------------------------------------------------
// att: a = exp(gate - max);  ssum[g] += a;  xatt[g,:] += a * x[n,:]
// (normalization by ssum deferred to head kernel)
// ---------------------------------------------------------------------------
__global__ void att_kernel(const float* __restrict__ x, const int* __restrict__ batch,
                           const float* __restrict__ gate, const unsigned* __restrict__ gmax,
                           float* __restrict__ ssum, float* __restrict__ xatt) {
  int n = blockIdx.x * blockDim.x + threadIdx.x;
  if (n >= NN) return;
  int g = batch[n];
  float a = __expf(gate[n] - dec_f(gmax[g]));
  atomicAdd(&ssum[g], a);
#pragma unroll
  for (int i = 0; i < 16; ++i) atomicAdd(&xatt[g * 16 + i], a * x[n * 16 + i]);
}

// ---------------------------------------------------------------------------
// conv1 messages: per wave = 16 edges.  A = (h (x) x_src | x_src | 0) f16,
// B = W21 (LDS), 9 k-steps x 2 n-tiles -> 18 wmma; scatter atomicAdd to agg1.
// Private-array indices kept compile-time by computing both lane-half variants.
// ---------------------------------------------------------------------------
#define FRAG1(K) ((K) < 256 ? (_Float16)(hrow[(K) >> 4] * xs[(K)&15]) \
                            : ((K) < 272 ? xs[(K)-256] : (_Float16)0.0f))

__global__ void __launch_bounds__(256)
conv1_kernel(const float* __restrict__ x, const float* __restrict__ ea,
             const float* __restrict__ w_e1a, const float* __restrict__ b_e1a,
             const int* __restrict__ ei, const _Float16* __restrict__ W21,
             float* __restrict__ agg1) {
  __shared__ __align__(32) _Float16 Bs[K1 * 32];   // column-major [32][288]
  {
    const uint4* s = (const uint4*)W21;
    uint4* d = (uint4*)Bs;
    for (int i = threadIdx.x; i < (K1 * 32) / 8; i += 256) d[i] = s[i];
  }
  __syncthreads();

  const int lane = threadIdx.x & 31;
  const int wave = threadIdx.x >> 5;
  const int m = lane & 15;
  const int hi = lane >> 4;
  const int base = (blockIdx.x * 8 + wave) * 16;

  const int e = base + m;
  const int e_cl = e < EE ? e : EE - 1;
  const float on = e < EE ? 1.0f : 0.0f;

  // edge MLP: h = relu(ea @ w_e1a + b)   [16], weights read scalar (uniform)
  float eav[8];
#pragma unroll
  for (int j = 0; j < 8; ++j) eav[j] = on * ea[e_cl * 8 + j];
  _Float16 hrow[16];
#pragma unroll
  for (int k = 0; k < 16; ++k) {
    float acc = b_e1a[k];
#pragma unroll
    for (int j = 0; j < 8; ++j) acc += eav[j] * w_e1a[j * 16 + k];
    hrow[k] = (_Float16)(on * fmaxf(acc, 0.0f));
  }
  // gather source node features
  const int srcn = ei[e_cl];
  _Float16 xs[16];
#pragma unroll
  for (int i = 0; i < 16; ++i) xs[i] = (_Float16)(on * x[srcn * 16 + i]);

  v8f c0 = {}, c1 = {};
#pragma unroll
  for (int s = 0; s < 9; ++s) {
    v16h a;
#pragma unroll
    for (int t = 0; t < 16; ++t) {
      const int Kb = s * 32 + t;   // compile-time
      _Float16 vlo = FRAG1(Kb);
      _Float16 vhi = FRAG1(Kb + 16);
      a[t] = hi ? vhi : vlo;
    }
    const int ko = s * 32 + hi * 16;
    v16h b0 = load_bcol(&Bs[m * K1 + ko]);
    v16h b1 = load_bcol(&Bs[(16 + m) * K1 + ko]);
    c0 = wmma16(a, b0, c0);
    c1 = wmma16(a, b1, c1);
  }
#pragma unroll
  for (int r = 0; r < 8; ++r) {
    const int eo = base + r + hi * 8;
    if (eo < EE) {
      const int tg = ei[EE + eo];
      atomicAdd(&agg1[tg * 32 + m], c0[r]);
      atomicAdd(&agg1[tg * 32 + 16 + m], c1[r]);
    }
  }
}

// ---------------------------------------------------------------------------
// node1: h1 = relu(x @ w_root1 + b_root1 + agg1), stored f16.  2 wmma / wave.
// ---------------------------------------------------------------------------
__global__ void __launch_bounds__(256)
node1_kernel(const float* __restrict__ x, const float* __restrict__ w_root1,
             const float* __restrict__ b_root1, const float* __restrict__ agg1,
             _Float16* __restrict__ h1) {
  const int lane = threadIdx.x & 31;
  const int wave = threadIdx.x >> 5;
  const int m = lane & 15;
  const int hi = lane >> 4;
  const int base = (blockIdx.x * 8 + wave) * 16;
  const int node = base + m;
  const int n_cl = node < NN ? node : NN - 1;
  const float on = node < NN ? 1.0f : 0.0f;

  v16h a, b0, b1;
#pragma unroll
  for (int t = 0; t < 16; ++t) {
    const int K = t + hi * 16;       // 0..31, K>=16 is pad
    a[t] = (K < 16) ? (_Float16)(on * x[n_cl * 16 + K]) : (_Float16)0.0f;
    const int Kb = t + hi * 16;
    b0[t] = (Kb < 16) ? (_Float16)w_root1[Kb * 32 + m] : (_Float16)0.0f;
    b1[t] = (Kb < 16) ? (_Float16)w_root1[Kb * 32 + 16 + m] : (_Float16)0.0f;
  }
  v8f c0 = {}, c1 = {};
  c0 = wmma16(a, b0, c0);
  c1 = wmma16(a, b1, c1);
#pragma unroll
  for (int r = 0; r < 8; ++r) {
    const int nn = base + r + hi * 8;
    if (nn < NN) {
      float v0 = fmaxf(c0[r] + b_root1[m] + agg1[nn * 32 + m], 0.0f);
      float v1 = fmaxf(c1[r] + b_root1[16 + m] + agg1[nn * 32 + 16 + m], 0.0f);
      h1[nn * 32 + m] = (_Float16)v0;
      h1[nn * 32 + 16 + m] = (_Float16)v1;
    }
  }
}

// ---------------------------------------------------------------------------
// conv2 messages: IN=32, OUT=16, K = 544 (17 k-steps x 1 n-tile = 17 wmma)
// ---------------------------------------------------------------------------
#define FRAG2(K) ((K) < 512 ? (_Float16)(hrow[(K) >> 5] * xs[(K)&31]) \
                            : ((K) < 544 ? xs[(K)-512] : (_Float16)0.0f))

__global__ void __launch_bounds__(256)
conv2_kernel(const _Float16* __restrict__ h1, const float* __restrict__ ea,
             const float* __restrict__ w_e2a, const float* __restrict__ b_e2a,
             const int* __restrict__ ei, const _Float16* __restrict__ W22,
             float* __restrict__ agg2) {
  __shared__ __align__(32) _Float16 Bs[K2 * 16];   // column-major [16][544]
  {
    const uint4* s = (const uint4*)W22;
    uint4* d = (uint4*)Bs;
    for (int i = threadIdx.x; i < (K2 * 16) / 8; i += 256) d[i] = s[i];
  }
  __syncthreads();

  const int lane = threadIdx.x & 31;
  const int wave = threadIdx.x >> 5;
  const int m = lane & 15;
  const int hi = lane >> 4;
  const int base = (blockIdx.x * 8 + wave) * 16;

  const int e = base + m;
  const int e_cl = e < EE ? e : EE - 1;
  const float on = e < EE ? 1.0f : 0.0f;
  const _Float16 on16 = (_Float16)on;

  float eav[8];
#pragma unroll
  for (int j = 0; j < 8; ++j) eav[j] = on * ea[e_cl * 8 + j];
  _Float16 hrow[16];
#pragma unroll
  for (int k = 0; k < 16; ++k) {
    float acc = b_e2a[k];
#pragma unroll
    for (int j = 0; j < 8; ++j) acc += eav[j] * w_e2a[j * 16 + k];
    hrow[k] = (_Float16)(on * fmaxf(acc, 0.0f));
  }
  const int srcn = ei[e_cl];
  _Float16 xs[32];
#pragma unroll
  for (int i = 0; i < 32; ++i) xs[i] = on16 * h1[srcn * 32 + i];

  v8f c0 = {};
#pragma unroll
  for (int s = 0; s < 17; ++s) {
    v16h a;
#pragma unroll
    for (int t = 0; t < 16; ++t) {
      const int Kb = s * 32 + t;
      _Float16 vlo = FRAG2(Kb);
      _Float16 vhi = FRAG2(Kb + 16);
      a[t] = hi ? vhi : vlo;
    }
    v16h b0 = load_bcol(&Bs[m * K2 + s * 32 + hi * 16]);
    c0 = wmma16(a, b0, c0);
  }
#pragma unroll
  for (int r = 0; r < 8; ++r) {
    const int eo = base + r + hi * 8;
    if (eo < EE) {
      const int tg = ei[EE + eo];
      atomicAdd(&agg2[tg * 16 + m], c0[r]);
    }
  }
}

// ---------------------------------------------------------------------------
// node2: h2 = relu(h1 @ w_root2 + b_root2 + agg2); accumulate mean-pool sums.
// ---------------------------------------------------------------------------
__global__ void __launch_bounds__(256)
node2_kernel(const _Float16* __restrict__ h1, const float* __restrict__ w_root2,
             const float* __restrict__ b_root2, const float* __restrict__ agg2,
             const int* __restrict__ batch, float* __restrict__ meansum) {
  const int lane = threadIdx.x & 31;
  const int wave = threadIdx.x >> 5;
  const int m = lane & 15;
  const int hi = lane >> 4;
  const int base = (blockIdx.x * 8 + wave) * 16;
  const int node = base + m;
  const int n_cl = node < NN ? node : NN - 1;
  const _Float16 on16 = (_Float16)(node < NN ? 1.0f : 0.0f);

  v16h a, b0;
#pragma unroll
  for (int t = 0; t < 16; ++t) {
    const int K = t + hi * 16;   // full K=32
    a[t] = on16 * h1[n_cl * 32 + K];
    const int Kb = t + hi * 16;
    b0[t] = (_Float16)w_root2[Kb * 16 + m];
  }
  v8f c0 = {};
  c0 = wmma16(a, b0, c0);
#pragma unroll
  for (int r = 0; r < 8; ++r) {
    const int nn = base + r + hi * 8;
    if (nn < NN) {
      float v = fmaxf(c0[r] + b_root2[m] + agg2[nn * 16 + m], 0.0f);
      atomicAdd(&meansum[batch[nn] * 16 + m], v);
    }
  }
}

// ---------------------------------------------------------------------------
// head: z = [meansum/cnt, xatt/ssum] @ w_l1 + b_l1 -> @ w_l2 + b_l2
// ---------------------------------------------------------------------------
__global__ void head_kernel(const float* __restrict__ meansum, const float* __restrict__ cnt,
                            const float* __restrict__ xatt, const float* __restrict__ ssum,
                            const float* __restrict__ w_l1, const float* __restrict__ b_l1,
                            const float* __restrict__ w_l2, const float* __restrict__ b_l2,
                            float* __restrict__ out) {
  int g = blockIdx.x * blockDim.x + threadIdx.x;
  if (g >= GG) return;
  float z[32];
  float c = fmaxf(cnt[g], 1.0f);
  float s = ssum[g];
  s = (s > 0.0f) ? s : 1.0f;
#pragma unroll
  for (int i = 0; i < 16; ++i) {
    z[i] = meansum[g * 16 + i] / c;
    z[16 + i] = xatt[g * 16 + i] / s;
  }
  float o = b_l2[0];
#pragma unroll
  for (int j = 0; j < 8; ++j) {
    float acc = b_l1[j];
#pragma unroll
    for (int i = 0; i < 32; ++i) acc += z[i] * w_l1[i * 8 + j];
    o += acc * w_l2[j];
  }
  out[g] = o;
}

// ---------------------------------------------------------------------------
// launch
// ---------------------------------------------------------------------------
extern "C" void kernel_launch(void* const* d_in, const int* in_sizes, int n_in,
                              void* d_out, int out_size, void* d_ws, size_t ws_size,
                              hipStream_t stream) {
  const float* x       = (const float*)d_in[0];
  const float* ea      = (const float*)d_in[1];
  const float* w_e1a   = (const float*)d_in[2];
  const float* b_e1a   = (const float*)d_in[3];
  const float* w_e1b   = (const float*)d_in[4];
  const float* b_e1b   = (const float*)d_in[5];
  const float* w_root1 = (const float*)d_in[6];
  const float* b_root1 = (const float*)d_in[7];
  const float* w_e2a   = (const float*)d_in[8];
  const float* b_e2a   = (const float*)d_in[9];
  const float* w_e2b   = (const float*)d_in[10];
  const float* b_e2b   = (const float*)d_in[11];
  const float* w_root2 = (const float*)d_in[12];
  const float* b_root2 = (const float*)d_in[13];
  const float* w_g1    = (const float*)d_in[14];
  const float* b_g1    = (const float*)d_in[15];
  const float* w_g2    = (const float*)d_in[16];
  const float* b_g2    = (const float*)d_in[17];
  const float* w_l1    = (const float*)d_in[18];
  const float* b_l1    = (const float*)d_in[19];
  const float* w_l2    = (const float*)d_in[20];
  const float* b_l2    = (const float*)d_in[21];
  const int*   ei      = (const int*)d_in[22];
  const int*   batch   = (const int*)d_in[23];
  float* out = (float*)d_out;

  // ---- workspace layout (zeroed region first, one memset) ----
  char* ws = (char*)d_ws;
  size_t off = 0;
  float*    agg1    = (float*)(ws + off); off += (size_t)NN * 32 * 4;
  float*    agg2    = (float*)(ws + off); off += (size_t)NN * 16 * 4;
  float*    meansum = (float*)(ws + off); off += (size_t)GG * 16 * 4;
  float*    xatt    = (float*)(ws + off); off += (size_t)GG * 16 * 4;
  float*    cnt     = (float*)(ws + off); off += (size_t)GG * 4;
  unsigned* gmax    = (unsigned*)(ws + off); off += (size_t)GG * 4;
  float*    ssum    = (float*)(ws + off); off += (size_t)GG * 4;
  const size_t zero_bytes = off;
  off = (off + 255) & ~(size_t)255;
  float*    gate = (float*)(ws + off); off += (size_t)NN * 4;
  off = (off + 255) & ~(size_t)255;
  _Float16* h1   = (_Float16*)(ws + off); off += (size_t)NN * 32 * 2;
  off = (off + 255) & ~(size_t)255;
  _Float16* W21  = (_Float16*)(ws + off); off += (size_t)K1 * 32 * 2;
  off = (off + 255) & ~(size_t)255;
  _Float16* W22  = (_Float16*)(ws + off); off += (size_t)K2 * 16 * 2;

  hipMemsetAsync(ws, 0, zero_bytes, stream);   // gmax=0 encodes "lowest float"

  prep_kernel<<<64, 256, 0, stream>>>(w_e1b, b_e1b, w_e2b, b_e2b, W21, W22);

  gate_kernel<<<(NN + 255) / 256, 256, 0, stream>>>(x, w_g1, b_g1, w_g2, b_g2,
                                                    batch, gate, gmax, cnt);
  att_kernel<<<(NN + 255) / 256, 256, 0, stream>>>(x, batch, gate, gmax, ssum, xatt);

  conv1_kernel<<<(EE + 127) / 128, 256, 0, stream>>>(x, ea, w_e1a, b_e1a, ei, W21, agg1);
  node1_kernel<<<(NN + 127) / 128, 256, 0, stream>>>(x, w_root1, b_root1, agg1, h1);
  conv2_kernel<<<(EE + 127) / 128, 256, 0, stream>>>(h1, ea, w_e2a, b_e2a, ei, W22, agg2);
  node2_kernel<<<(NN + 127) / 128, 256, 0, stream>>>(h1, w_root2, b_root2, agg2,
                                                     batch, meansum);

  head_kernel<<<(GG + 255) / 256, 256, 0, stream>>>(meansum, cnt, xatt, ssum,
                                                    w_l1, b_l1, w_l2, b_l2, out);
}
